// AGNN_62663572848799
// MI455X (gfx1250) — matrix operand order repaired and checked by reading, not compiled, not run on hardware
//
#include <hip/hip_runtime.h>
#include <hip/hip_bf16.h>

typedef __attribute__((ext_vector_type(16))) _Float16 v16h;
typedef __attribute__((ext_vector_type(8)))  float    v8f;

#define DH 64                     // hidden width (reference: 64), 2 floats/lane in wave32
#define ORD_NEG ((int)0x80000000) // identity for order-preserving-int max

// order-preserving float<->int map (involution): any neg-encoded < any nonneg-encoded
__device__ __forceinline__ int   f2ord(float f) { int i = __float_as_int(f); return i >= 0 ? i : (i ^ 0x7fffffff); }
__device__ __forceinline__ float ord2f(int i)   { return __int_as_float(i >= 0 ? i : (i ^ 0x7fffffff)); }

// ---------------------------------------------------------------------------
// Pack W1 [DIN x 64] f32 -> f16 in WMMA B-fragment layout:
//   pB[((kt*4 + ct)*32 + lane)*16 + i] = W1[(kt*32 + 8*(lane>>4) + i + (i&8)) * 64 + ct*16 + (lane&15)]
// so a wave's whole 16-f16 B fragment is one aligned 32-byte load.
__global__ void pack_W1_f16(const float* __restrict__ W1, _Float16* __restrict__ pB, int DIN) {
  int t = blockIdx.x * blockDim.x + threadIdx.x;
  int total = (DIN >> 5) * 4 * 32 * 16;
  if (t >= total) return;
  int i    = t & 15;
  int lane = (t >> 4) & 31;
  int ct   = (t >> 9) & 3;
  int kt   = t >> 11;
  int k = kt * 32 + 8 * (lane >> 4) + i + (i & 8);
  int n = ct * 16 + (lane & 15);
  pB[t] = (_Float16)W1[(size_t)k * DH + n];
}

// Pack W4 [64 x C] (C<=48) padded to 48 cols: 2 k-tiles x 3 col-tiles
//   pB[((kt*3 + ct)*32 + lane)*16 + i]
__global__ void pack_W4_f16(const float* __restrict__ W4, _Float16* __restrict__ pB, int C) {
  int t = blockIdx.x * blockDim.x + threadIdx.x;
  int total = 2 * 3 * 32 * 16;
  if (t >= total) return;
  int i    = t & 15;
  int lane = (t >> 4) & 31;
  int rest = t >> 9;               // kt*3 + ct
  int ct = rest % 3, kt = rest / 3;
  int k = kt * 32 + 8 * (lane >> 4) + i + (i & 8);
  int n = ct * 16 + (lane & 15);
  pB[t] = (n < C) ? (_Float16)W4[(size_t)k * C + n] : (_Float16)0.f;
}

// ---------------------------------------------------------------------------
// GEMM1: h = relu(x @ W1 + b1)  [N x DIN] * [DIN x 64] -> [N x 64]
// block = 128 (4 waves); wave w owns the 16-col tile [16w,16w+16).
// A converted f32->f16 in flight; B fragment = single v16h load from pB.
__global__ void agnn_gemm1_wmma(const float* __restrict__ x, const _Float16* __restrict__ pB,
                                const float* __restrict__ b1, float* __restrict__ h,
                                int N, int DIN) {
  const int wave = threadIdx.x >> 5;
  const int lane = threadIdx.x & 31;
  const int half = lane >> 4;          // A: K block 0..7/16..23 vs 8..15/24..31
  const int r    = lane & 15;
  const int row0 = blockIdx.x * 16;
  const int n    = wave * 16 + r;
  const int rr   = min(row0 + r, N - 1);
  const float* xrow = x + (size_t)rr * DIN + 8 * half;
  const v16h* bfrag = (const v16h*)pB + (size_t)wave * 32 + lane;  // + kt*128
  v8f acc = {};
#pragma unroll 4
  for (int kt = 0; kt < (DIN >> 5); ++kt) {
    const float* xp = xrow + kt * 32;
    float4 a0 = *(const float4*)(xp + 0);
    float4 a1 = *(const float4*)(xp + 4);
    float4 a2 = *(const float4*)(xp + 16);
    float4 a3 = *(const float4*)(xp + 20);
    v16h A;
    A[0]=(_Float16)a0.x;  A[1]=(_Float16)a0.y;  A[2]=(_Float16)a0.z;  A[3]=(_Float16)a0.w;
    A[4]=(_Float16)a1.x;  A[5]=(_Float16)a1.y;  A[6]=(_Float16)a1.z;  A[7]=(_Float16)a1.w;
    A[8]=(_Float16)a2.x;  A[9]=(_Float16)a2.y;  A[10]=(_Float16)a2.z; A[11]=(_Float16)a2.w;
    A[12]=(_Float16)a3.x; A[13]=(_Float16)a3.y; A[14]=(_Float16)a3.z; A[15]=(_Float16)a3.w;
    v16h B = bfrag[(size_t)kt * 128];
    acc = __builtin_amdgcn_wmma_f32_16x16x32_f16(false, A, false, B, (short)0, acc, false, false);
  }
  const float bias = b1[n];
#pragma unroll
  for (int i = 0; i < 8; ++i) {        // C/D: VGPR i -> M = i + 8*half, N = lane&15
    int m = row0 + i + 8 * half;
    if (m < N) {
      float v = acc[i] + bias;
      h[(size_t)m * DH + n] = v > 0.f ? v : 0.f;
    }
  }
}

// ---------------------------------------------------------------------------
// GEMM2 + log_softmax: out = log_softmax(h @ W4 + b4)  [N x 64]*[64 x C], C<=48
// block = 96 (3 waves -> 48 padded cols); logits staged in LDS, then 16
// threads do the per-row log-softmax.
__global__ void agnn_gemm2_lsm_wmma(const float* __restrict__ h, const _Float16* __restrict__ pB,
                                    const float* __restrict__ b4, float* __restrict__ out,
                                    int N, int C) {
  __shared__ float lds[16 * 48];
  const int wave = threadIdx.x >> 5;
  const int lane = threadIdx.x & 31;
  const int half = lane >> 4;
  const int r    = lane & 15;
  const int row0 = blockIdx.x * 16;
  const int n    = wave * 16 + r;
  const int rr   = min(row0 + r, N - 1);
  const float* hrow = h + (size_t)rr * DH + 8 * half;
  const v16h* bfrag = (const v16h*)pB + (size_t)wave * 32 + lane;  // + kt*96
  v8f acc = {};
#pragma unroll
  for (int kt = 0; kt < 2; ++kt) {
    const float* xp = hrow + kt * 32;
    float4 a0 = *(const float4*)(xp + 0);
    float4 a1 = *(const float4*)(xp + 4);
    float4 a2 = *(const float4*)(xp + 16);
    float4 a3 = *(const float4*)(xp + 20);
    v16h A;
    A[0]=(_Float16)a0.x;  A[1]=(_Float16)a0.y;  A[2]=(_Float16)a0.z;  A[3]=(_Float16)a0.w;
    A[4]=(_Float16)a1.x;  A[5]=(_Float16)a1.y;  A[6]=(_Float16)a1.z;  A[7]=(_Float16)a1.w;
    A[8]=(_Float16)a2.x;  A[9]=(_Float16)a2.y;  A[10]=(_Float16)a2.z; A[11]=(_Float16)a2.w;
    A[12]=(_Float16)a3.x; A[13]=(_Float16)a3.y; A[14]=(_Float16)a3.z; A[15]=(_Float16)a3.w;
    v16h B = bfrag[(size_t)kt * 96];
    acc = __builtin_amdgcn_wmma_f32_16x16x32_f16(false, A, false, B, (short)0, acc, false, false);
  }
  const float bias = (n < C) ? b4[n] : 0.f;
#pragma unroll
  for (int i = 0; i < 8; ++i)
    lds[(i + 8 * half) * 48 + n] = acc[i] + bias;
  __syncthreads();
  if (threadIdx.x < 16) {
    int row = row0 + (int)threadIdx.x;
    if (row < N) {
      const float* rp = lds + threadIdx.x * 48;
      float mx = -3.402823466e38f;
      for (int j = 0; j < C; ++j) mx = fmaxf(mx, rp[j]);
      float s = 0.f;
      for (int j = 0; j < C; ++j) s += __expf(rp[j] - mx);
      float lse = mx + __logf(s);
      float* o = out + (size_t)row * C;
      for (int j = 0; j < C; ++j) o[j] = rp[j] - lse;
    }
  }
}

// ---------------------------------------------------------------------------
// Row L2-normalize: one wave per node (64 floats = 1 float2/lane)
__global__ void agnn_normalize(const float* __restrict__ h, float* __restrict__ hn, int N) {
  int node = blockIdx.x * (blockDim.x >> 5) + (threadIdx.x >> 5);
  if (node >= N) return;
  int lane = threadIdx.x & 31;
  float2 v = *(const float2*)(h + (size_t)node * DH + 2 * lane);
  float ss = v.x * v.x + v.y * v.y;
#pragma unroll
  for (int off = 16; off; off >>= 1) ss += __shfl_xor(ss, off, 32);
  float inv = 1.f / fmaxf(sqrtf(ss), 1e-12f);
  float2 o; o.x = v.x * inv; o.y = v.y * inv;
  *(float2*)(hn + (size_t)node * DH + 2 * lane) = o;
}

// Per-edge attention logit + segment max (order-preserving int atomicMax).
// One wave per edge; id >= E is the self-loop (id - E).
__global__ void agnn_edge_e_max(const int* __restrict__ ei, int E, int ET,
                                const float* __restrict__ hn, const float* __restrict__ betas,
                                int li, float* __restrict__ ebuf, int* __restrict__ menc) {
  int id = blockIdx.x * (blockDim.x >> 5) + (threadIdx.x >> 5);
  if (id >= ET) return;
  int lane = threadIdx.x & 31;
  int s, d;
  if (id < E) { s = ei[id]; d = ei[E + id]; } else { s = d = id - E; }
  float2 a = *(const float2*)(hn + (size_t)s * DH + 2 * lane);
  float2 b = *(const float2*)(hn + (size_t)d * DH + 2 * lane);
  float dot = a.x * b.x + a.y * b.y;
#pragma unroll
  for (int off = 16; off; off >>= 1) dot += __shfl_xor(dot, off, 32);
  float e = betas[li] * dot;
  if (lane == 0) {
    ebuf[id] = e;
    atomicMax(menc + d, f2ord(e));
  }
}

// ee = exp(e - m[dst]); denom[dst] += ee.  One thread per edge.
__global__ void agnn_edge_exp(const int* __restrict__ ei, int E, int ET,
                              float* __restrict__ ebuf, const int* __restrict__ menc,
                              float* __restrict__ denom) {
  int id = blockIdx.x * blockDim.x + threadIdx.x;
  if (id >= ET) return;
  int d = (id < E) ? ei[E + id] : (id - E);
  int me = menc[d];
  float m = (me == ORD_NEG) ? 0.f : ord2f(me);   // ref: non-finite max -> 0
  float ee = __expf(ebuf[id] - m);
  ebuf[id] = ee;
  atomicAdd(denom + d, ee);
}

// hout[dst] += (ee/denom[dst]) * h[src].  One wave per edge, f32 atomics.
__global__ void agnn_edge_scatter(const int* __restrict__ ei, int E, int ET,
                                  const float* __restrict__ ebuf, const float* __restrict__ denom,
                                  const float* __restrict__ h, float* __restrict__ hout) {
  int id = blockIdx.x * (blockDim.x >> 5) + (threadIdx.x >> 5);
  if (id >= ET) return;
  int lane = threadIdx.x & 31;
  int s, d;
  if (id < E) { s = ei[id]; d = ei[E + id]; } else { s = d = id - E; }
  float alpha = ebuf[id] / fmaxf(denom[d], 1e-12f);
  float2 v = *(const float2*)(h + (size_t)s * DH + 2 * lane);
  float* po = hout + (size_t)d * DH + 2 * lane;
  atomicAdd(po,     alpha * v.x);
  atomicAdd(po + 1, alpha * v.y);
}

__global__ void fill_f32(float* __restrict__ p, float v, long long n) {
  long long i = (long long)blockIdx.x * blockDim.x + threadIdx.x;
  if (i < n) p[i] = v;
}
__global__ void fill_i32(int* __restrict__ p, int v, long long n) {
  long long i = (long long)blockIdx.x * blockDim.x + threadIdx.x;
  if (i < n) p[i] = v;
}

static inline size_t alignup(size_t x) { return (x + 255) & ~(size_t)255; }

extern "C" void kernel_launch(void* const* d_in, const int* in_sizes, int n_in,
                              void* d_out, int out_size, void* d_ws, size_t ws_size,
                              hipStream_t stream) {
  const float* x     = (const float*)d_in[0];
  const int*   ei    = (const int*)d_in[1];   // [2, E] row-major
  const float* W1    = (const float*)d_in[2];
  const float* b1    = (const float*)d_in[3];
  const float* W4    = (const float*)d_in[4];
  const float* b4    = (const float*)d_in[5];
  const float* betas = (const float*)d_in[6];

  const int DHl = in_sizes[3];            // 64
  const int C   = in_sizes[5];            // 40
  const int DIN = in_sizes[2] / DHl;      // 512
  const int N   = in_sizes[0] / DIN;      // 50000
  const int E   = in_sizes[1] / 2;        // 800000
  const int ET  = E + N;                  // edges + self-loops
  (void)DHl; (void)n_in; (void)out_size; (void)ws_size;

  // workspace: h0/h1/hn (N*64 f32), ebuf (ET f32), menc (N i32), denom (N f32),
  // packed weights pB1 (DIN*64 f16) + pB4 (3072 f16)  ~ 42.3 MB total
  char* w = (char*)d_ws;
  float*     h0    = (float*)w;     w += alignup((size_t)N * DH * 4);
  float*     h1    = (float*)w;     w += alignup((size_t)N * DH * 4);
  float*     hn    = (float*)w;     w += alignup((size_t)N * DH * 4);
  float*     ebuf  = (float*)w;     w += alignup((size_t)ET * 4);
  int*       menc  = (int*)w;       w += alignup((size_t)N * 4);
  float*     denom = (float*)w;     w += alignup((size_t)N * 4);
  _Float16*  pB1   = (_Float16*)w;  w += alignup((size_t)DIN * DH * 2);
  _Float16*  pB4   = (_Float16*)w;

  const int rowTiles = (N + 15) / 16;
  const long long nh = (long long)N * DH;
  const int nPack1 = (DIN >> 5) * 4 * 32 * 16;
  const int nPack4 = 2 * 3 * 32 * 16;

  // pre-pack weights into WMMA B-fragment layout (f16)
  pack_W1_f16<<<(nPack1 + 255) / 256, 256, 0, stream>>>(W1, pB1, DIN);
  pack_W4_f16<<<(nPack4 + 255) / 256, 256, 0, stream>>>(W4, pB4, C);

  // h = relu(x @ W1 + b1)   (WMMA)
  agnn_gemm1_wmma<<<rowTiles, 128, 0, stream>>>(x, pB1, b1, h0, N, DIN);

  float* hcur = h0;
  float* hnxt = h1;
  for (int l = 0; l < 4; ++l) {
    fill_i32<<<(N + 255) / 256, 256, 0, stream>>>(menc, ORD_NEG, N);
    fill_f32<<<(N + 255) / 256, 256, 0, stream>>>(denom, 0.f, N);
    fill_f32<<<(int)((nh + 255) / 256), 256, 0, stream>>>(hnxt, 0.f, nh);
    agnn_normalize<<<(N + 7) / 8, 256, 0, stream>>>(hcur, hn, N);
    agnn_edge_e_max<<<(ET + 7) / 8, 256, 0, stream>>>(ei, E, ET, hn, betas, l, ebuf, menc);
    agnn_edge_exp<<<(ET + 255) / 256, 256, 0, stream>>>(ei, E, ET, ebuf, menc, denom);
    agnn_edge_scatter<<<(ET + 7) / 8, 256, 0, stream>>>(ei, E, ET, ebuf, denom, hcur, hnxt);
    float* t = hcur; hcur = hnxt; hnxt = t;
  }

  // out = log_softmax(h @ W4 + b4)   (WMMA + LDS softmax)
  agnn_gemm2_lsm_wmma<<<rowTiles, 96, 0, stream>>>(hcur, pB4, b4, (float*)d_out, N, C);
}
